// Forward_2662879723663
// MI455X (gfx1250) — matrix operand order repaired and checked by reading, not compiled, not run on hardware
//
#include <hip/hip_runtime.h>
#include <hip/hip_bf16.h>
#include <math.h>

typedef __attribute__((ext_vector_type(8)))  _Float16 half8;
typedef __attribute__((ext_vector_type(16))) _Float16 v16h;
typedef __attribute__((ext_vector_type(8)))  float    v8f;

#define B_SZ   2048
#define IN_SZ  32
#define H_SZ   1024
#define NOSC   16
#define NSPEC  1024

// ---------------------------------------------------------------------------
// WMMA fragment helpers (CDNA5 16x16x32 f16, wave32)
// A layout (ISA 7.12.2): lane m=L&15, kg=L>>4; halves 0..7 -> K = k0+8*kg+0..7,
//                        halves 8..15 -> K = k0+16+8*kg+0..7  (row-major A, stride K)
// B layout: weights pre-transposed to N-major (row n holds all K); lane n=L&15,
//           kg=L>>4; halves 0..7 -> K = k0+8*kg+0..7, halves 8..15 -> K = k0+16+8*kg+0..7
// C/D layout: lane n=L&15, mh=L>>4; element i -> row M0+i+8*mh, col N0+n
// ---------------------------------------------------------------------------
__device__ __forceinline__ v16h load_a_frag(const _Float16* __restrict__ A,
                                            int row0, int K, int k0, int lane) {
    int m  = lane & 15;
    int kg = lane >> 4;
    const _Float16* p = A + (size_t)(row0 + m) * K + k0 + kg * 8;
    half8 lo = *(const half8*)(p);
    half8 hi = *(const half8*)(p + 16);
    v16h a;
#pragma unroll
    for (int i = 0; i < 8; ++i) { a[i] = lo[i]; a[i + 8] = hi[i]; }
    return a;
}

__device__ __forceinline__ v16h load_b_frag(const _Float16* __restrict__ BT,
                                            int col0, int K, int k0, int lane) {
    int n  = lane & 15;
    int kg = lane >> 4;
    const _Float16* p = BT + (size_t)(col0 + n) * K + k0 + kg * 8;
    half8 lo = *(const half8*)(p);
    half8 hi = *(const half8*)(p + 16);
    v16h b;
#pragma unroll
    for (int i = 0; i < 8; ++i) { b[i] = lo[i]; b[i + 8] = hi[i]; }
    return b;
}

__device__ __forceinline__ v8f wmma_f16(v16h a, v16h b, v8f c) {
    return __builtin_amdgcn_wmma_f32_16x16x32_f16(false, a, false, b,
                                                  (short)0, c, false, false);
}

__device__ __forceinline__ void store_tile_relu_f16(v8f c, _Float16* __restrict__ O,
                                                    int N, int row0, int col0,
                                                    const float* __restrict__ bias,
                                                    int lane) {
    int n  = lane & 15;
    int mh = lane >> 4;
    float bv = bias[col0 + n];
#pragma unroll
    for (int i = 0; i < 8; ++i) {
        float v = c[i] + bv;
        v = v > 0.0f ? v : 0.0f;
        O[(size_t)(row0 + i + 8 * mh) * N + col0 + n] = (_Float16)v;
    }
}

// ---------------------------------------------------------------------------
// GEMM + bias + ReLU, one wave = 32x32 output (2x2 WMMA tiles).
// Used for layer 1 (K=32, single K-step; maximize wave count).
// ---------------------------------------------------------------------------
__global__ void gemm_bias_relu_f16_t32(const _Float16* __restrict__ A,
                                       const _Float16* __restrict__ BT,
                                       const float* __restrict__ bias,
                                       _Float16* __restrict__ O,
                                       int M, int N, int K) {
    int lane = threadIdx.x & 31;
    int wave = (blockIdx.x * blockDim.x + threadIdx.x) >> 5;
    int n2   = N >> 5;
    int tm   = wave / n2;
    int tn   = wave - tm * n2;
    int M0 = tm << 5;
    int N0 = tn << 5;
    if (M0 >= M) return;

    v8f c00 = {}, c01 = {}, c10 = {}, c11 = {};
    for (int k0 = 0; k0 < K; k0 += 32) {
        v16h a0 = load_a_frag(A, M0,      K, k0, lane);
        v16h a1 = load_a_frag(A, M0 + 16, K, k0, lane);
        v16h b0 = load_b_frag(BT, N0,      K, k0, lane);
        v16h b1 = load_b_frag(BT, N0 + 16, K, k0, lane);
        c00 = wmma_f16(a0, b0, c00);
        c01 = wmma_f16(a0, b1, c01);
        c10 = wmma_f16(a1, b0, c10);
        c11 = wmma_f16(a1, b1, c11);
    }
    store_tile_relu_f16(c00, O, N, M0,      N0,      bias, lane);
    store_tile_relu_f16(c01, O, N, M0,      N0 + 16, bias, lane);
    store_tile_relu_f16(c10, O, N, M0 + 16, N0,      bias, lane);
    store_tile_relu_f16(c11, O, N, M0 + 16, N0 + 16, bias, lane);
}

// ---------------------------------------------------------------------------
// GEMM + bias + ReLU, one wave = 64x64 output (4x4 WMMA tiles).
// 16 WMMA per 16 b128 loads per K-step: double the fragment reuse of the
// 32x32 version. Used for the two compute-dominant 1024x1024 layers.
// ---------------------------------------------------------------------------
__global__ void gemm_bias_relu_f16_t64(const _Float16* __restrict__ A,
                                       const _Float16* __restrict__ BT,
                                       const float* __restrict__ bias,
                                       _Float16* __restrict__ O,
                                       int M, int N, int K) {
    int lane = threadIdx.x & 31;
    int wave = (blockIdx.x * blockDim.x + threadIdx.x) >> 5;
    int n4   = N >> 6;
    int tm   = wave / n4;
    int tn   = wave - tm * n4;
    int M0 = tm << 6;
    int N0 = tn << 6;
    if (M0 >= M) return;

    v8f acc[4][4] = {};
#pragma unroll 1
    for (int k0 = 0; k0 < K; k0 += 32) {
        v16h a[4], b[4];
#pragma unroll
        for (int i = 0; i < 4; ++i) a[i] = load_a_frag(A, M0 + 16 * i, K, k0, lane);
#pragma unroll
        for (int j = 0; j < 4; ++j) b[j] = load_b_frag(BT, N0 + 16 * j, K, k0, lane);
#pragma unroll
        for (int i = 0; i < 4; ++i) {
#pragma unroll
            for (int j = 0; j < 4; ++j) {
                acc[i][j] = wmma_f16(a[i], b[j], acc[i][j]);
            }
        }
    }
#pragma unroll
    for (int i = 0; i < 4; ++i) {
#pragma unroll
        for (int j = 0; j < 4; ++j) {
            store_tile_relu_f16(acc[i][j], O, N, M0 + 16 * i, N0 + 16 * j, bias, lane);
        }
    }
}

// ---------------------------------------------------------------------------
// Head GEMM: h(2048x1024) @ WheadT(48x1024)^T, leaky-ReLU, f32 out.
// One wave per 16x16 tile; tileN (0..2) selects w0 / wp / g output.
// ---------------------------------------------------------------------------
__global__ void head_gemm(const _Float16* __restrict__ A,
                          const _Float16* __restrict__ WT,
                          const float* __restrict__ bias48,
                          float* __restrict__ w0,
                          float* __restrict__ wp,
                          float* __restrict__ g) {
    int lane = threadIdx.x & 31;
    int wave = (blockIdx.x * blockDim.x + threadIdx.x) >> 5;
    int tm = wave / 3;           // 0..127
    int tn = wave - tm * 3;      // 0..2 (head index)
    if (tm >= (B_SZ / 16)) return;
    int M0 = tm << 4;
    int N0 = tn << 4;

    v8f c = {};
    for (int k0 = 0; k0 < H_SZ; k0 += 32) {
        v16h a = load_a_frag(A, M0, H_SZ, k0, lane);
        v16h b = load_b_frag(WT, N0, H_SZ, k0, lane);
        c = wmma_f16(a, b, c);
    }
    float* O = (tn == 0) ? w0 : (tn == 1) ? wp : g;
    int n  = lane & 15;
    int mh = lane >> 4;
    float bv = bias48[N0 + n];
#pragma unroll
    for (int i = 0; i < 8; ++i) {
        float v = c[i] + bv;
        v = v > 0.0f ? v : 0.01f * v;   // jax leaky_relu default slope
        O[(size_t)(M0 + i + 8 * mh) * NOSC + n] = v;
    }
}

// ---------------------------------------------------------------------------
// Spectral kernel: thread per (batch, freq); oscillator params in LDS.
// ---------------------------------------------------------------------------
__global__ void spectral_kernel(const float* __restrict__ w0,
                                const float* __restrict__ wp,
                                const float* __restrict__ g,
                                const float* __restrict__ dvec,
                                float* __restrict__ T) {
    __shared__ float sw0[NOSC], swp[NOSC], sg[NOSC];
    __shared__ float sd;
    int b  = blockIdx.x >> 2;
    int kb = (blockIdx.x & 3) << 8;
    int t  = threadIdx.x;
    if (t < NOSC) {
        sw0[t] = w0[b * NOSC + t];
        swp[t] = wp[b * NOSC + t];
        sg[t]  = g[b * NOSC + t];
    }
    if (t == 0) sd = dvec[b];
    __syncthreads();

    int k = kb + t;
    float f  = 0.5f + (4.5f / 1024.0f) * (float)k;
    float w2 = f * f;
    float e1 = 5.0f;   // EPS_INF folded in
    float e2 = 0.0f;
#pragma unroll
    for (int o = 0; o < NOSC; ++o) {
        float wp2  = swp[o] * swp[o];
        float gb   = sg[o] * 0.1f;
        float diff = sw0[o] * sw0[o] - w2;
        float den  = diff * diff + w2 * gb * gb;
        float inv  = 1.0f / den;
        e1 += wp2 * diff * inv;
        e2 += wp2 * f * gb * inv;
    }
    // principal complex sqrt of (e1 + i*e2)
    float r  = sqrtf(e1 * e1 + e2 * e2);
    float nr = sqrtf(fmaxf(0.5f * (r + e1), 0.0f));
    float ni = copysignf(sqrtf(fmaxf(0.5f * (r - e1), 0.0f)), e2);
    float alpha = expf(-0.006283185307179586f * sd * ni); // -0.0005*4*pi
    float denom = (nr + 1.0f) * (nr + 1.0f) + ni * ni;
    float Tv = 4.0f * nr / denom * alpha;
    T[(size_t)b * NSPEC + k] = Tv;
}

// ---------------------------------------------------------------------------
// Prep kernels
// ---------------------------------------------------------------------------
__global__ void cvt_f32_to_f16(const float* __restrict__ in,
                               _Float16* __restrict__ out, int n) {
    int i = blockIdx.x * blockDim.x + threadIdx.x;
    if (i < n) out[i] = (_Float16)in[i];
}

// in: f32 row-major [K x N] -> out: f16 row-major [N x K]  (i.e. transposed)
__global__ void transpose_to_f16(const float* __restrict__ in,
                                 _Float16* __restrict__ out, int K, int N) {
    int i = blockIdx.x * blockDim.x + threadIdx.x;
    if (i >= K * N) return;
    int k = i / N;
    int n = i - k * N;
    out[(size_t)n * K + k] = (_Float16)in[i];
}

// Pack Ww0/Wwp/Wg (each H x 16, row-major) into WheadT f16 [48 x H]
__global__ void build_head_wt(const float* __restrict__ Ww0,
                              const float* __restrict__ Wwp,
                              const float* __restrict__ Wg,
                              _Float16* __restrict__ out) {
    int i = blockIdx.x * blockDim.x + threadIdx.x;
    if (i >= 48 * H_SZ) return;
    int n = i / H_SZ;      // 0..47
    int k = i - n * H_SZ;  // 0..1023
    int head = n >> 4;
    int oc   = n & 15;
    const float* src = (head == 0) ? Ww0 : (head == 1) ? Wwp : Wg;
    out[(size_t)n * H_SZ + k] = (_Float16)src[(size_t)k * NOSC + oc];
}

__global__ void build_bias48(const float* __restrict__ bw0,
                             const float* __restrict__ bwp,
                             const float* __restrict__ bg,
                             float* __restrict__ out) {
    int t = threadIdx.x;
    if (t < NOSC) {
        out[t]          = bw0[t];
        out[NOSC + t]   = bwp[t];
        out[2*NOSC + t] = bg[t];
    }
}

// d[b] = max(G[b,16:32]) * 5 + 47
__global__ void d_reduce(const float* __restrict__ G, float* __restrict__ dvec) {
    int b = blockIdx.x * blockDim.x + threadIdx.x;
    if (b >= B_SZ) return;
    float m = -1.0e30f;
#pragma unroll
    for (int j = 0; j < NOSC; ++j)
        m = fmaxf(m, G[(size_t)b * IN_SZ + NOSC + j]);
    dvec[b] = m * 5.0f + 47.0f;
}

// ---------------------------------------------------------------------------
extern "C" void kernel_launch(void* const* d_in, const int* in_sizes, int n_in,
                              void* d_out, int out_size, void* d_ws, size_t ws_size,
                              hipStream_t stream) {
    const float* G   = (const float*)d_in[0];
    const float* W1  = (const float*)d_in[1];
    const float* b1  = (const float*)d_in[2];
    const float* W2  = (const float*)d_in[3];
    const float* b2  = (const float*)d_in[4];
    const float* W3  = (const float*)d_in[5];
    const float* b3  = (const float*)d_in[6];
    const float* Ww0 = (const float*)d_in[7];
    const float* bw0 = (const float*)d_in[8];
    const float* Wwp = (const float*)d_in[9];
    const float* bwp = (const float*)d_in[10];
    const float* Wg  = (const float*)d_in[11];
    const float* bg  = (const float*)d_in[12];

    char* ws = (char*)d_ws;
    size_t off = 0;
    auto alloc = [&](size_t bytes) -> void* {
        void* p = ws + off;
        off = (off + bytes + 255) & ~(size_t)255;
        return p;
    };
    _Float16* Gf16  = (_Float16*)alloc((size_t)B_SZ * IN_SZ * 2);
    _Float16* W1T   = (_Float16*)alloc((size_t)H_SZ * IN_SZ * 2);
    _Float16* W2T   = (_Float16*)alloc((size_t)H_SZ * H_SZ * 2);
    _Float16* W3T   = (_Float16*)alloc((size_t)H_SZ * H_SZ * 2);
    _Float16* WhT   = (_Float16*)alloc((size_t)48 * H_SZ * 2);
    float*    bias48= (float*)alloc(48 * 4);
    float*    dvec  = (float*)alloc((size_t)B_SZ * 4);
    _Float16* H1    = (_Float16*)alloc((size_t)B_SZ * H_SZ * 2);
    _Float16* H2    = (_Float16*)alloc((size_t)B_SZ * H_SZ * 2);
    _Float16* H3    = (_Float16*)alloc((size_t)B_SZ * H_SZ * 2);

    float* Tout = (float*)d_out;                          // [2048 x 1024]
    float* w0o  = Tout + (size_t)B_SZ * NSPEC;            // [2048 x 16]
    float* wpo  = w0o + (size_t)B_SZ * NOSC;
    float* go   = wpo + (size_t)B_SZ * NOSC;

    // Prep
    cvt_f32_to_f16<<<(B_SZ * IN_SZ + 255) / 256, 256, 0, stream>>>(G, Gf16, B_SZ * IN_SZ);
    transpose_to_f16<<<(IN_SZ * H_SZ + 255) / 256, 256, 0, stream>>>(W1, W1T, IN_SZ, H_SZ);
    transpose_to_f16<<<(H_SZ * H_SZ + 255) / 256, 256, 0, stream>>>(W2, W2T, H_SZ, H_SZ);
    transpose_to_f16<<<(H_SZ * H_SZ + 255) / 256, 256, 0, stream>>>(W3, W3T, H_SZ, H_SZ);
    build_head_wt<<<(48 * H_SZ + 255) / 256, 256, 0, stream>>>(Ww0, Wwp, Wg, WhT);
    build_bias48<<<1, 64, 0, stream>>>(bw0, bwp, bg, bias48);
    d_reduce<<<(B_SZ + 255) / 256, 256, 0, stream>>>(G, dvec);

    // Layer 1: K=32, single K-step -> 32x32 tiles, 2048 waves, 256 blocks
    {
        int waves  = (B_SZ / 32) * (H_SZ / 32);
        int blocks = waves / 8;
        gemm_bias_relu_f16_t32<<<blocks, 256, 0, stream>>>(Gf16, W1T, b1, H1,
                                                           B_SZ, H_SZ, IN_SZ);
    }
    // Layers 2 & 3: 64x64 tiles per wave, 512 waves, 64 blocks
    {
        int waves  = (B_SZ / 64) * (H_SZ / 64);
        int blocks = waves / 8;
        gemm_bias_relu_f16_t64<<<blocks, 256, 0, stream>>>(H1, W2T, b2, H2,
                                                           B_SZ, H_SZ, H_SZ);
        gemm_bias_relu_f16_t64<<<blocks, 256, 0, stream>>>(H2, W3T, b3, H3,
                                                           B_SZ, H_SZ, H_SZ);
    }

    // Heads: 128 M-tiles * 3 N-tiles = 384 waves -> 48 blocks
    head_gemm<<<48, 256, 0, stream>>>(H3, WhT, bias48, w0o, wpo, go);

    // Spectral: one block per (batch, 256-freq chunk)
    spectral_kernel<<<B_SZ * (NSPEC / 256), 256, 0, stream>>>(w0o, wpo, go, dvec, Tout);
}
